// MetabolicModel_16862041604654
// MI455X (gfx1250) — matrix operand order repaired and checked by reading, not compiled, not run on hardware
//
#include <hip/hip_runtime.h>
#include <cmath>

// ---------------- problem constants ----------------
#define BB   512     // batch
#define TT   100     // time steps
#define KK   50      // concentrations
#define NN   2712    // fluxes
#define MM   1877    // metabolites
#define HH   500     // hidden
// padded dims (tile-friendly)
#define KP   64      // K padded to 2 k-chunks
#define HP   512     // H padded
#define NP   2752    // N padded (64*43, also 32*86)
#define MP   1920    // M padded (64*30)

typedef __attribute__((ext_vector_type(8)))  float  v8f;
typedef __attribute__((ext_vector_type(8)))  __bf16 v8bf;
typedef __attribute__((ext_vector_type(16))) __bf16 v16bf;
typedef __attribute__((ext_vector_type(4)))  unsigned int u32x4;
typedef __attribute__((ext_vector_type(8)))  int i32x8;
typedef __attribute__((ext_vector_type(4)))  int i32x4;

#define LDSR 40                    // padded LDS row stride (bf16): 64B data + 16B pad
#define BM   256                   // block rows
#define ABUF_BYTES (BM * LDSR * 2)
#define BBUF_BYTES (64 * LDSR * 2)

#define HAS_TDM __has_builtin(__builtin_amdgcn_tensor_load_to_lds)

// ---------------- prep kernels ----------------
__global__ void k_pad_direct(const float* __restrict__ src, __bf16* __restrict__ dst,
                             int srcRows, int srcCols, int dstRows, int dstCols) {
    int idx = blockIdx.x * 256 + threadIdx.x;
    int total = dstRows * dstCols;
    if (idx >= total) return;
    int r = idx / dstCols, c = idx % dstCols;
    float v = (r < srcRows && c < srcCols) ? src[(size_t)r * srcCols + c] : 0.f;
    dst[idx] = (__bf16)v;
}

__global__ void k_pad_transpose(const float* __restrict__ src, __bf16* __restrict__ dst,
                                int srcRows, int srcCols, int dstRows, int dstCols) {
    // dst[r][c] = src[c][r], zero-padded
    int idx = blockIdx.x * 256 + threadIdx.x;
    int total = dstRows * dstCols;
    if (idx >= total) return;
    int r = idx / dstCols, c = idx % dstCols;
    float v = (r < srcCols && c < srcRows) ? src[(size_t)c * srcCols + r] : 0.f;
    dst[idx] = (__bf16)v;
}

__global__ void k_init(const float* __restrict__ cref, float* __restrict__ Cf,
                       __bf16* __restrict__ Cb, float* __restrict__ acc) {
    int idx = blockIdx.x * 256 + threadIdx.x;
    if (idx < 5 * TT) acc[idx] = 0.f;
    if (idx >= BB * KP) return;
    int b = idx >> 6, k = idx & 63;
    float v = (k < KK) ? cref[((size_t)b * (TT + 1)) * KK + k] : 0.f;  // C_ref[b][0][k]
    Cb[idx] = (__bf16)v;
    if (k < KK) Cf[b * KK + k] = v;
}

// ---------------- helpers ----------------
__device__ inline float waveRed(float v) {
#pragma unroll
    for (int o = 16; o > 0; o >>= 1) v += __shfl_xor(v, o, 32);
    return v;
}

// A fragment (16x32 bf16): half0 -> K[0..7],[16..23]; half1 -> K[8..15],[24..31]
__device__ inline v16bf load_frag_a(const __bf16* row, int half) {
    v8bf lo = *(const v8bf*)(row + 8 * half);
    v8bf hi = *(const v8bf*)(row + 16 + 8 * half);
    v16bf r;
#pragma unroll
    for (int i = 0; i < 8; ++i) { r[i] = lo[i]; r[i + 8] = hi[i]; }
    return r;
}
// B fragment (32x16 bf16): half0 -> K[0..7],[16..23]; half1 -> K[8..15],[24..31]
__device__ inline v16bf load_frag_b(const __bf16* row, int half) {
    v8bf lo = *(const v8bf*)(row + 8 * half);
    v8bf hi = *(const v8bf*)(row + 16 + 8 * half);
    v16bf r;
#pragma unroll
    for (int i = 0; i < 8; ++i) { r[i] = lo[i]; r[i + 8] = hi[i]; }
    return r;
}

// ---------------- Tensor Data Mover descriptors (D#) ----------------
// Group 0: [1:0]=count(1), [63:32]=lds_addr, [120:64]=global_addr, [127:126]=type(2)
__device__ inline u32x4 tdm_group0(unsigned lds_addr, unsigned long long gaddr) {
    u32x4 g0;
    g0[0] = 1u;
    g0[1] = lds_addr;
    g0[2] = (unsigned)(gaddr & 0xFFFFFFFFull);
    g0[3] = (unsigned)((gaddr >> 32) & 0x1FFFFFFull) | (2u << 30);
    return g0;
}
// Group 1: data_size=1 (2B), pad_enable, pad_interval=3 (16 DW = 64B), pad_amount=3 (4 DW = 16B)
// tensor_dim0 = tile_w (no OOB, padded buffers), tensor_dim1 = tile_h,
// tensor_dim0_stride = row stride (elements)
__device__ inline i32x8 tdm_group1(unsigned tile_w, unsigned tile_h,
                                   unsigned long long row_stride_elems) {
    i32x8 g1;
    g1[0] = (int)((1u << 16) | (1u << 20) | (3u << 22) | (3u << 25));
    g1[1] = (int)((tile_w & 0xFFFFu) << 16);                          // tensor_dim0 lo16
    g1[2] = (int)((tile_w >> 16) | ((tile_h & 0xFFFFu) << 16));       // dim0 hi16 | dim1 lo16
    g1[3] = (int)((tile_h >> 16) | (tile_w << 16));                   // dim1 hi16 | tile_dim0
    g1[4] = (int)(tile_h & 0xFFFFu);                                  // tile_dim1 | tile_dim2=0
    g1[5] = (int)(row_stride_elems & 0xFFFFFFFFull);                  // dim0_stride lo32
    g1[6] = (int)((row_stride_elems >> 32) & 0xFFFFull);              // dim0_stride hi16
    g1[7] = 0;
    return g1;
}

__device__ inline void tdm_issue(u32x4 g0, i32x8 g1) {
#if HAS_TDM
    i32x4 z4 = {0, 0, 0, 0};
#if defined(__clang_major__) && (__clang_major__ >= 23)
    i32x8 z8 = {0, 0, 0, 0, 0, 0, 0, 0};
    __builtin_amdgcn_tensor_load_to_lds(g0, g1, z4, z4, z8, 0);
#else
    __builtin_amdgcn_tensor_load_to_lds(g0, g1, z4, z4, 0);
#endif
#else
    (void)g0; (void)g1;
#endif
}

// ---------------- fused WMMA GEMM (TDM double-buffered) ----------------
// Block tile 256x64, 8 waves stacked along M; each wave computes a 32x64 strip:
// per 32-K chunk -> 2 A-frags + 4 B-frags (12 ds_load_b128) feeding 8 WMMAs.
// MODE 0: h = relu(C@W1 + b1)            -> outBF (bf16)
// MODE 1: v = h@W2 + b2                  -> outBF, accumulate sum(relu(-v)) -> s0
// MODE 2: C += (v@T^T)*dt, masked MSE vs ref -> s0(diff2) s1(cnt) s2(drop)
// MODE 3: sum((v@S^T)^2)                 -> s0 (no store)
template <int MODE>
__global__ __launch_bounds__(256) void gemm_wmma_k(
    const __bf16* __restrict__ A, int lda,
    const __bf16* __restrict__ Bm, int ldb,
    int kIters, int realN, int ldout,
    __bf16* __restrict__ outBF,
    const float* __restrict__ bias,
    float* __restrict__ Cstate,
    const float* __restrict__ cref,
    const float* __restrict__ dtvec,
    int t,
    float* __restrict__ s0, float* __restrict__ s1, float* __restrict__ s2)
{
    __shared__ __align__(16) __bf16 sA[2 * BM * LDSR];
    __shared__ __align__(16) __bf16 sB[2 * 64 * LDSR];

    const int tid  = threadIdx.x;
    const int lane = tid & 31;
    const int wv   = tid >> 5;        // 8 waves, stacked along M
    const int half = lane >> 4;
    const int l15  = lane & 15;

    const int mBase = blockIdx.y * BM;
    const int nBase = blockIdx.x * 64;

#if HAS_TDM
    // Uniform TDM descriptors (SGPR); only global_addr changes per K-chunk (+64B).
    const unsigned ldsA0 = (unsigned)(unsigned long long)(const void*)&sA[0];
    const unsigned ldsB0 = (unsigned)(unsigned long long)(const void*)&sB[0];
    const unsigned long long gA = (unsigned long long)(A + (size_t)mBase * lda);
    const unsigned long long gB = (unsigned long long)(Bm + (size_t)nBase * ldb);
    const i32x8 g1A = tdm_group1(32u, (unsigned)BM, (unsigned long long)lda);
    const i32x8 g1B = tdm_group1(32u, 64u, (unsigned long long)ldb);
#else
    const __bf16* aSrcRow = A  + (size_t)(mBase + tid) * lda;          // full row per thread
    const int brow = tid >> 2, bseg = tid & 3;
    const __bf16* bSrcRow = Bm + (size_t)(nBase + brow) * ldb + bseg * 8;
#endif

    v8f accs[2][4];
#pragma unroll
    for (int i = 0; i < 2; ++i)
#pragma unroll
        for (int j = 0; j < 4; ++j) accs[i][j] = (v8f){0,0,0,0,0,0,0,0};

#if HAS_TDM
    if (wv == 0) {  // prime buffer 0
        tdm_issue(tdm_group0(ldsA0, gA), g1A);
        tdm_issue(tdm_group0(ldsB0, gB), g1B);
    }
#endif

    for (int kb = 0; kb < kIters; ++kb) {
        const int p = kb & 1;
#if HAS_TDM
        if (wv == 0) {
            if (kb + 1 < kIters) {   // issue next chunk into other buffer, wait for current
                unsigned long long koff = (unsigned long long)(kb + 1) * 64ull;  // 32 bf16
                tdm_issue(tdm_group0(ldsA0 + (1 - p) * ABUF_BYTES, gA + koff), g1A);
                tdm_issue(tdm_group0(ldsB0 + (1 - p) * BBUF_BYTES, gB + koff), g1B);
                __builtin_amdgcn_s_wait_tensorcnt((short)2);
            } else {
                __builtin_amdgcn_s_wait_tensorcnt((short)0);
            }
        }
#else
        {   // fallback: cooperative copy into buffer p
            const u32x4* as = (const u32x4*)(aSrcRow + kb * 32);
            u32x4* ad = (u32x4*)(&sA[p * BM * LDSR + tid * LDSR]);
            ad[0] = as[0];
            ad[1] = as[1];
            *(u32x4*)(&sB[p * 64 * LDSR + brow * LDSR + bseg * 8]) =
                *(const u32x4*)(bSrcRow + kb * 32);
        }
#endif
        __syncthreads();   // buffer p ready (wave0 observed TENSORcnt, others gated here)

        const __bf16* sAp = sA + p * BM * LDSR;
        const __bf16* sBp = sB + p * 64 * LDSR;
        v16bf af[2], bf[4];
#pragma unroll
        for (int fm = 0; fm < 2; ++fm)
            af[fm] = load_frag_a(sAp + (wv * 32 + fm * 16 + l15) * LDSR, half);
#pragma unroll
        for (int fn = 0; fn < 4; ++fn)
            bf[fn] = load_frag_b(sBp + (fn * 16 + l15) * LDSR, half);

#pragma unroll
        for (int fm = 0; fm < 2; ++fm)
#pragma unroll
            for (int fn = 0; fn < 4; ++fn)
                accs[fm][fn] = __builtin_amdgcn_wmma_f32_16x16x32_bf16(
                    false, af[fm], false, bf[fn], (short)0, accs[fm][fn], false, false);

        __syncthreads();   // all waves done reading buffer p; it may be overwritten
    }

    // ---------------- epilogue ----------------
    float local0 = 0.f, local1 = 0.f, local2 = 0.f;
    float dt = 0.f;
    if constexpr (MODE == 2) dt = dtvec[t];

#pragma unroll
    for (int fm = 0; fm < 2; ++fm) {
#pragma unroll
        for (int fn = 0; fn < 4; ++fn) {
            v8f acc = accs[fm][fn];
#pragma unroll
            for (int r = 0; r < 8; ++r) {
                int gm = mBase + wv * 32 + fm * 16 + half * 8 + r;  // C/D: M = r + 8*half
                int gn = nBase + fn * 16 + l15;                     // C/D: N = lane&15
                float val = acc[r];
                if constexpr (MODE == 0) {
                    float o = 0.f;
                    if (gn < realN) o = fmaxf(val + bias[gn], 0.f);
                    outBF[(size_t)gm * ldout + gn] = (__bf16)o;
                } else if constexpr (MODE == 1) {
                    float o = 0.f;
                    if (gn < realN) { o = val + bias[gn]; local0 += fmaxf(-o, 0.f); }
                    outBF[(size_t)gm * ldout + gn] = (__bf16)o;
                } else if constexpr (MODE == 2) {
                    if (gn < realN) {
                        float cold = Cstate[gm * KK + gn];
                        float cnew = cold + val * dt;
                        Cstate[gm * KK + gn] = cnew;
                        outBF[(size_t)gm * ldout + gn] = (__bf16)cnew;
                        float ref = cref[((size_t)gm * (TT + 1) + (t + 1)) * KK + gn];
                        if (!(ref != ref)) { float d = cnew - ref; local0 += d * d; local1 += 1.f; }
                        if (gn == KK - 1) local2 += fmaxf(cold - cnew, 0.f);
                    }
                } else {
                    local0 += val * val;   // sum(S_v^2); padded rows contribute exactly 0
                }
            }
        }
    }
    if constexpr (MODE == 1 || MODE == 2 || MODE == 3) {
        local0 = waveRed(local0);
        if constexpr (MODE == 2) { local1 = waveRed(local1); local2 = waveRed(local2); }
        if (lane == 0) {
            atomicAdd(s0, local0);
            if constexpr (MODE == 2) { atomicAdd(s1, local1); atomicAdd(s2, local2); }
        }
    }
}

// ---------------- final loss assembly ----------------
__global__ void k_final(const float* __restrict__ acc, float* __restrict__ out) {
    if (blockIdx.x != 0 || threadIdx.x != 0) return;
    // acc layout: [0]=sv_sum [1]=negv_sum [2]=diff2 [3]=cnt [4]=drop, each length TT
    float ls = 0.f, lc = 0.f;
    for (int t = 0; t < TT; ++t) {
        float ts = (float)t;
        float w = expf(-0.5f * ts);
        float mse_sv = acc[0 * TT + t] / ((float)BB * (float)MM)
                     + (acc[1 * TT + t] / ((float)BB * (float)NN)) * w;
        ls += mse_sv * 0.001f;
        float cnt = fmaxf(acc[3 * TT + t], 1.0f);
        float mse_c = acc[2 * TT + t] / cnt + (acc[4 * TT + t] / (float)BB) * w;
        lc += mse_c * w;
    }
    out[0] = ls;
    out[1] = lc;
}

// ---------------- host launcher ----------------
extern "C" void kernel_launch(void* const* d_in, const int* in_sizes, int n_in,
                              void* d_out, int out_size, void* d_ws, size_t ws_size,
                              hipStream_t stream)
{
    const float* Cref = (const float*)d_in[0];   // (512,101,50)
    const float* dtv  = (const float*)d_in[1];   // (100,)
    const float* W1   = (const float*)d_in[2];   // (50,500)
    const float* b1   = (const float*)d_in[3];   // (500,)
    const float* W2   = (const float*)d_in[4];   // (500,2712)
    const float* b2   = (const float*)d_in[5];   // (2712,)
    const float* Tr   = (const float*)d_in[6];   // (50,2712)
    const float* St   = (const float*)d_in[7];   // (1877,2712)
    float* out = (float*)d_out;

    char* ws = (char*)d_ws;
    size_t off = 0;
    auto take = [&](size_t bytes) -> char* {
        char* p = ws + off;
        off = (off + bytes + 255) & ~(size_t)255;
        return p;
    };
    __bf16* W1T = (__bf16*)take((size_t)HP * KP * 2);   // [512][64]
    __bf16* W2T = (__bf16*)take((size_t)NP * HP * 2);   // [2752][512]
    __bf16* TrB = (__bf16*)take((size_t)KP * NP * 2);   // [64][2752]
    __bf16* StB = (__bf16*)take((size_t)MP * NP * 2);   // [1920][2752]
    float*  Cf  = (float*) take((size_t)BB * KK * 4);   // f32 state
    __bf16* Cb  = (__bf16*)take((size_t)BB * KP * 2);   // bf16 state (padded)
    __bf16* Hb  = (__bf16*)take((size_t)BB * HP * 2);
    __bf16* Vb  = (__bf16*)take((size_t)BB * NP * 2);
    float*  acc = (float*) take((size_t)5 * TT * 4);
    (void)ws_size; (void)in_sizes; (void)n_in; (void)out_size;

    auto g1 = [](long long n) { return dim3((unsigned)((n + 255) / 256)); };
    k_pad_transpose<<<g1((long long)HP * KP), 256, 0, stream>>>(W1, W1T, KK, HH, HP, KP);
    k_pad_transpose<<<g1((long long)NP * HP), 256, 0, stream>>>(W2, W2T, HH, NN, NP, HP);
    k_pad_direct  <<<g1((long long)KP * NP), 256, 0, stream>>>(Tr, TrB, KK, NN, KP, NP);
    k_pad_direct  <<<g1((long long)MP * NP), 256, 0, stream>>>(St, StB, MM, NN, MP, NP);
    k_init        <<<g1((long long)BB * KP), 256, 0, stream>>>(Cref, Cf, Cb, acc);

    for (int t = 0; t < TT; ++t) {
        // h = relu(C @ W1 + b1)
        gemm_wmma_k<0><<<dim3(HP / 64, BB / BM), 256, 0, stream>>>(
            Cb, KP, W1T, KP, KP / 32, HH, HP, Hb, b1,
            nullptr, nullptr, nullptr, t, nullptr, nullptr, nullptr);
        // v = h @ W2 + b2  (+ relu(-v) reduction)
        gemm_wmma_k<1><<<dim3(NP / 64, BB / BM), 256, 0, stream>>>(
            Hb, HP, W2T, HP, HP / 32, NN, NP, Vb, b2,
            nullptr, nullptr, nullptr, t, acc + 1 * TT + t, nullptr, nullptr);
        // C += (v @ T^T) * dt  (+ masked MSE / drop reductions)
        gemm_wmma_k<2><<<dim3(KP / 64, BB / BM), 256, 0, stream>>>(
            Vb, NP, TrB, NP, NP / 32, KK, KP, Cb, nullptr,
            Cf, Cref, dtv, t, acc + 2 * TT + t, acc + 3 * TT + t, acc + 4 * TT + t);
        // sum((v @ S^T)^2)  (no store)
        gemm_wmma_k<3><<<dim3(MP / 64, BB / BM), 256, 0, stream>>>(
            Vb, NP, StB, NP, NP / 32, MM, 0, nullptr, nullptr,
            nullptr, nullptr, nullptr, t, acc + 0 * TT + t, nullptr, nullptr);
    }
    k_final<<<1, 1, 0, stream>>>(acc, out);
}
